// LIF_R_ASC_31387620999431
// MI455X (gfx1250) — compile-verified, matrix operand order checked by reading, not run on hardware
//
#include <hip/hip_runtime.h>
#include <hip/hip_bf16.h>

// ---------------------------------------------------------------------------
// LIF recurrent network, persistent-kernel formulation for MI455X (gfx1250).
//
// w (4096x4096 fp32, 64MB) would be re-read from L2 on every one of the 2048
// timesteps (128 GB of traffic) in a streaming design. Instead each of 128
// workgroups pins its 32-column slice of w into the 320KB WGP LDS as f16 in
// WMMA-B register layout, once. Per step only the 8KB I_add vector is
// broadcast from L2 (~1MB/step device-wide). The vec-mat uses
// v_wmma_f32_16x16x32_f16 with the I_add vector in row 0 of the A tile
// (fp32 accumulation; C row 0 depends only on A row 0, so other rows are
// don't-care). Neuron state lives in wave-0 registers for the whole run.
// One device-wide phased atomic barrier per timestep orders the recurrence.
//
// Inner loop is software-pipelined: iteration it+1's ds_load_b128s are issued
// before iteration it's WMMAs consume their data, and one A load is shared by
// the two column-tile WMMAs (6 instead of 8 b128 loads per 2 WMMAs).
// ---------------------------------------------------------------------------

typedef __attribute__((ext_vector_type(16))) _Float16 v16h;
typedef __attribute__((ext_vector_type(8)))  _Float16 v8h;
typedef __attribute__((ext_vector_type(8)))  float    v8f;
typedef __attribute__((ext_vector_type(4)))  unsigned int uint4v;

#define NN       4096
#define TSTEPS   2048
#define NWG      128
#define COLS_WG  32      // neurons (w columns) per workgroup: 2 tiles of 16
#define THREADS  256     // 8 wave32s

// LDS layout (bytes): w tile (2 x 128KB) | I staging (8KB) | reduction (1KB)
#define LDS_SI_OFF     (2 * 131072)           // 262144
#define LDS_RED_OFF    (LDS_SI_OFF + 8192)    // 270336
#define LDS_TOTAL      (LDS_RED_OFF + 2 * 8 * 16 * 4)  // 271360 B <= 320KB

__global__ void init_counter_kernel(int* cnt) {
  if (threadIdx.x == 0) *cnt = 0;   // fresh barrier counter every call (graph-replay safe)
}

__device__ __forceinline__ v16h cat16(v8h lo, v8h hi) {
  return __builtin_shufflevector(lo, hi, 0,1,2,3,4,5,6,7,8,9,10,11,12,13,14,15);
}

__global__ void lif_persistent_kernel(const float* __restrict__ x_in,
                                      const float* __restrict__ w,
                                      const float* __restrict__ E_L,
                                      const float* __restrict__ tau_m,
                                      const float* __restrict__ G,
                                      const float* __restrict__ R_I,
                                      const float* __restrict__ f_v,
                                      const float* __restrict__ f_I,
                                      const float* __restrict__ dth_s,
                                      const float* __restrict__ b_s,
                                      const float* __restrict__ d_V,
                                      const float* __restrict__ I_A,
                                      float* __restrict__ out,
                                      int* __restrict__ cnt,
                                      _Float16* __restrict__ ib0,
                                      _Float16* __restrict__ ib1) {
  extern __shared__ char smem[];
  _Float16* wT  = (_Float16*)(smem);                 // WMMA-B layout f16 tiles
  _Float16* sI  = (_Float16*)(smem + LDS_SI_OFF);    // staged I_add (f16)
  float*    red = (float*)(smem + LDS_RED_OFF);      // red[2 tiles][8 waves][16]

  const int tid  = threadIdx.x;
  const int wid  = tid >> 5;        // wave id 0..7 (wave32)
  const int lane = tid & 31;
  const int j0   = blockIdx.x * COLS_WG;

  // ---- one-time preload: w[:, j0:j0+32] -> LDS, f16, WMMA-B register image.
  // 16-bit B (32x16) layout mirrors the documented 16-bit A layout:
  //   lane = col + 16*h where h=(K>>3)&1 ; half index e=(K&7)+8*(K>>4)
  for (int idx = tid; idx < NN * COLS_WG; idx += THREADS) {
    const int c32 = idx & 31;
    const int r   = idx >> 5;                    // global row (K) 0..4095
    const float wv = w[(size_t)r * NN + j0 + c32];
    const int tt = c32 >> 4, ct = c32 & 15;
    const int K = r & 31, kb = r >> 5;
    const int h = (K >> 3) & 1;
    const int e = (K & 7) + ((K >> 4) << 3);
    wT[(size_t)tt * 65536 + kb * 512 + (ct + (h << 4)) * 16 + e] = (_Float16)wv;
  }
  __syncthreads();

  // ---- per-neuron state in wave-0 registers (32 neurons <-> threads 0..31)
  float vS = 0.f, thS = 0.f, Icur = 0.f;
  float eL = 0.f, tm = 1.f, g = 0.f, rI = 0.f, fv = 0.f, fI = 0.f;
  float dth = 0.f, bs = 0.f, dV = 0.f, iA = 0.f, wdiag = 0.f;
  const int jj = j0 + tid;
  if (tid < 32) {
    eL = E_L[jj];  tm = tau_m[jj]; g  = G[jj];     rI = R_I[jj];
    fv = f_v[jj];  fI = f_I[jj];   dth = dth_s[jj]; bs = b_s[jj];
    dV = d_V[jj];  iA = I_A[jj];
    vS = eL; thS = dth; Icur = 0.f;
    ib0[jj] = (_Float16)0.f;                     // I_add(t=0) = 0
    // self-weight (for exact diagonal cancellation of the WMMA dot)
    const int tt = tid >> 4, ct = tid & 15;
    const int K = jj & 31, kb = jj >> 5;
    const int h = (K >> 3) & 1;
    const int e = (K & 7) + ((K >> 4) << 3);
    wdiag = (float)wT[(size_t)tt * 65536 + kb * 512 + (ct + (h << 4)) * 16 + e];
    __builtin_prefetch(x_in + jj, 0, 0);         // first x row -> caches
    __threadfence();                              // publish ib0 before barrier
  }

  // ---- device-wide phased barrier (counter zeroed by init kernel each call)
  auto gbar = [&](int phase) {
    __syncthreads();
    if (tid == 0) {
      __hip_atomic_fetch_add(cnt, 1, __ATOMIC_RELEASE, __HIP_MEMORY_SCOPE_AGENT);
      const int tgt = NWG * (phase + 1);
      while (__hip_atomic_load(cnt, __ATOMIC_ACQUIRE, __HIP_MEMORY_SCOPE_AGENT) < tgt)
        __builtin_amdgcn_s_sleep(1);
    }
    __syncthreads();
  };

  gbar(0);  // all ib0 slices + LDS w tiles ready everywhere

  // per-wave LDS base pointers for the pipelined WMMA loop
  const int h = lane >> 4;
  const _Float16* aBase  = sI + wid * 512 + 8 * h;                   // +32 halfs/it
  const _Float16* b0Base = wT + (size_t)wid * (16 * 512) + lane * 16; // +512 halfs/it
  const _Float16* b1Base = b0Base + 65536;                            // tile 1

  for (int t = 0; t < TSTEPS; ++t) {
    const _Float16* icur = (t & 1) ? ib1 : ib0;
    _Float16*       inxt = (t & 1) ? ib0 : ib1;

    // stage I_add (4096 f16 = 8KB) from L2 into LDS, 128-bit moves
    {
      const uint4v* src = (const uint4v*)icur;
      uint4v*       dst = (uint4v*)sI;
      for (int i = tid; i < (NN * 2) / 16; i += THREADS) dst[i] = src[i];
    }
    __syncthreads();

    // ---- WMMA: wave `wid` covers K rows [wid*512, wid*512+512); both column
    // tiles share one A load; 2-stage software pipeline hides DS latency.
    v8f acc0 = {0.f,0.f,0.f,0.f,0.f,0.f,0.f,0.f};
    v8f acc1 = {0.f,0.f,0.f,0.f,0.f,0.f,0.f,0.f};
    v16h Ac  = cat16(*(const v8h*)(aBase),           *(const v8h*)(aBase + 16));
    v16h B0c = cat16(*(const v8h*)(b0Base),          *(const v8h*)(b0Base + 8));
    v16h B1c = cat16(*(const v8h*)(b1Base),          *(const v8h*)(b1Base + 8));
    #pragma unroll 5
    for (int it = 0; it < 15; ++it) {
      const _Float16* an = aBase  + (it + 1) * 32;
      const _Float16* bn = b0Base + (it + 1) * 512;
      const _Float16* cn = b1Base + (it + 1) * 512;
      const v16h An  = cat16(*(const v8h*)(an), *(const v8h*)(an + 16));
      const v16h B0n = cat16(*(const v8h*)(bn), *(const v8h*)(bn + 8));
      const v16h B1n = cat16(*(const v8h*)(cn), *(const v8h*)(cn + 8));
      acc0 = __builtin_amdgcn_wmma_f32_16x16x32_f16(false, Ac, false, B0c,
                                                    (short)0, acc0, false, false);
      acc1 = __builtin_amdgcn_wmma_f32_16x16x32_f16(false, Ac, false, B1c,
                                                    (short)0, acc1, false, false);
      Ac = An; B0c = B0n; B1c = B1n;
    }
    acc0 = __builtin_amdgcn_wmma_f32_16x16x32_f16(false, Ac, false, B0c,
                                                  (short)0, acc0, false, false);
    acc1 = __builtin_amdgcn_wmma_f32_16x16x32_f16(false, Ac, false, B1c,
                                                  (short)0, acc1, false, false);
    if (lane < 16) {                         // C row M=0 lives in c[0], lanes 0..15
      red[(0 * 8 + wid) * 16 + lane] = acc0[0];
      red[(1 * 8 + wid) * 16 + lane] = acc1[0];
    }
    __syncthreads();

    // ---- neuron update (wave 0): reduce 8 wave partials, exact reference math
    if (tid < 32) {
      const int tt = tid >> 4, ct = tid & 15;
      float dot = 0.f;
      #pragma unroll
      for (int wv = 0; wv < 8; ++wv) dot += red[(tt * 8 + wv) * 16 + ct];
      dot -= wdiag * (float)sI[jj];                       // zero the diagonal
      const float Iin = dot + 0.9f * x_in[(size_t)t * NN + jj];
      const float vn  = vS + (g * (eL - vS) + Iin * rI) / tm;
      out[(size_t)t * NN + jj] = 1.f / (1.f + __expf(-(vn - thS)));
      const bool sp = (vn >= thS);
      const float vr = eL + fv * (vS - eL) - dV;
      vS   = sp ? vr : vn;
      thS  = sp ? (thS + dth) : (thS - bs * thS);
      Icur = sp ? (Icur + iA) : ((1.f - fI) * Icur);
      inxt[jj] = (_Float16)Icur;                          // publish I_add(t+1)
      if (t + 1 < TSTEPS)                                  // hide next x_in row
        __builtin_prefetch(x_in + (size_t)(t + 1) * NN + jj, 0, 0);
      __threadfence();
    }
    gbar(t + 1);  // one barrier/step suffices with double-buffered I_add
  }
}

extern "C" void kernel_launch(void* const* d_in, const int* in_sizes, int n_in,
                              void* d_out, int out_size, void* d_ws, size_t ws_size,
                              hipStream_t stream) {
  const float* x_in = (const float*)d_in[0];
  const float* w    = (const float*)d_in[1];
  const float* E_L  = (const float*)d_in[2];
  const float* taum = (const float*)d_in[3];
  const float* G    = (const float*)d_in[4];
  const float* R_I  = (const float*)d_in[5];
  const float* f_v  = (const float*)d_in[6];
  const float* f_I  = (const float*)d_in[7];
  const float* dths = (const float*)d_in[8];
  const float* b_s  = (const float*)d_in[9];
  const float* d_V  = (const float*)d_in[10];
  const float* I_A  = (const float*)d_in[11];
  float* out = (float*)d_out;

  // workspace: barrier counter + double-buffered f16 I_add vectors
  int*      cnt = (int*)d_ws;
  _Float16* ib0 = (_Float16*)((char*)d_ws + 128);
  _Float16* ib1 = ib0 + NN;

  // allow >64KB dynamic LDS (ignore result; harmless if unsupported)
  (void)hipFuncSetAttribute((const void*)lif_persistent_kernel,
                            hipFuncAttributeMaxDynamicSharedMemorySize, LDS_TOTAL);

  init_counter_kernel<<<1, 32, 0, stream>>>(cnt);
  lif_persistent_kernel<<<NWG, THREADS, LDS_TOTAL, stream>>>(
      x_in, w, E_L, taum, G, R_I, f_v, f_I, dths, b_s, d_V, I_A,
      out, cnt, ib0, ib1);
}
